// MoChA_91207925498599
// MI455X (gfx1250) — matrix operand order repaired and checked by reading, not compiled, not run on hardware
//
#include <hip/hip_runtime.h>

// ---------------- constants ----------------
#define B_SZ   16
#define K_SZ   2000
#define Q_SZ   256
#define D_SZ   1024
#define ADIM_SZ 1024
#define H_SZ   4          // H_MA * H_CA (H_CA == 1)
#define DK_SZ  256
#define EPS_F  1e-6f
#define NEG_F  -1000000000.0f
#define SCALE_F 32.0f     // sqrt(1024)
#define SHARP_F 1.0f
#define WWIN   8

// ---------------- types ----------------
typedef __attribute__((ext_vector_type(16))) __bf16 v16bf;
typedef __attribute__((ext_vector_type(8)))  float  v8f;
typedef __attribute__((ext_vector_type(4)))  float  f32x4;
typedef __attribute__((ext_vector_type(4)))  unsigned short u16x4;

union Frag16 { v16bf v; unsigned short u[16]; };

__device__ __forceinline__ unsigned short f2bf(float x) {
  unsigned int u = __float_as_uint(x);
  u += 0x7FFFu + ((u >> 16) & 1u);   // round to nearest even
  return (unsigned short)(u >> 16);
}

__device__ __forceinline__ u16x4 f2bf4(f32x4 f) {
  u16x4 h;
  h.x = f2bf(f.x); h.y = f2bf(f.y); h.z = f2bf(f.z); h.w = f2bf(f.w);
  return h;
}

// A-fragment gather (16x32 bf16, M x K) from LDS tile, row stride lda (ushorts)
__device__ __forceinline__ void load_afrag(Frag16& a, const unsigned short* As,
                                           int lda, int mrow0, int lane) {
  int m  = mrow0 + (lane & 15);
  int kb = (lane >> 4) * 16;
  const unsigned short* row = As + m * lda;
#pragma unroll
  for (int e = 0; e < 16; ++e) {
    int k = kb + e;
    a.u[e] = row[k];
  }
}

// B-fragment gather (32x16 bf16, K x N) from LDS tile Bs[k][n]
__device__ __forceinline__ void load_bfrag(Frag16& b, const unsigned short* Bs,
                                           int ldb, int ncol0, int lane) {
  int n  = ncol0 + (lane & 15);
  int kb = (lane >> 4) * 16;
#pragma unroll
  for (int e = 0; e < 16; ++e) b.u[e] = Bs[(kb + e) * ldb + n];
}

// B-fragment gather TRANSPOSED: staged tile T[n][k]
__device__ __forceinline__ void load_bfrag_t(Frag16& b, const unsigned short* Ts,
                                             int ldt, int ncol0, int lane) {
  int n  = ncol0 + (lane & 15);
  int kb = (lane >> 4) * 16;
  const unsigned short* row = Ts + n * ldt;
#pragma unroll
  for (int e = 0; e < 16; ++e) b.u[e] = row[kb + e];
}

#define WMMA_BF16(acc, a, b) \
  acc = __builtin_amdgcn_wmma_f32_16x16x32_bf16(false, (a).v, false, (b).v, (short)0, acc, false, false)

// =====================================================================
// Kernel 1: C[M,N] = A[M,Dd] @ W[Dd,N] + bias.  Guard-free: M%128==0,
// N%64==0, Dd%32==0 (true for every projection call).
// block 256 = 8 waves; tile 128(M) x 64(N); wave (wm 0..3, wn 0..1) owns 32x32.
// =====================================================================
template <bool A_BF16, bool OUT_BF16>
__global__ void __launch_bounds__(256)
gemm_proj_kernel(const void* __restrict__ Av, const float* __restrict__ Wt,
                 const float* __restrict__ bias, void* __restrict__ Cv,
                 int M, int Dd, int N) {
  __shared__ __attribute__((aligned(16))) unsigned short As[128][36];
  __shared__ __attribute__((aligned(16))) unsigned short Bs[32][68];
  const int tid = threadIdx.x, lane = tid & 31, wave = tid >> 5;
  const int wm = wave & 3, wn = wave >> 2;
  const int m0 = blockIdx.x * 128, n0 = blockIdx.y * 64;
  const float* Af = (const float*)Av;
  const unsigned short* Ab = (const unsigned short*)Av;

  v8f acc00 = {}; v8f acc01 = {}; v8f acc10 = {}; v8f acc11 = {};

  for (int kk = 0; kk < Dd; kk += 32) {
    // stage A: 128x32, vec4 along k: 1024 vecs -> 4 per thread
#pragma unroll
    for (int v = tid; v < 128 * 8; v += 256) {
      int r = v >> 3, c4 = (v & 7) * 4;
      size_t g = (size_t)(m0 + r) * Dd + (kk + c4);
      u16x4 hv;
      if (A_BF16) {
        hv = *(const u16x4*)(Ab + g);
        if (kk + 32 < Dd) __builtin_prefetch((const void*)(Ab + g + 32), 0, 1);
      } else {
        f32x4 f = *(const f32x4*)(Af + g);
        hv = f2bf4(f);
        if (kk + 32 < Dd) __builtin_prefetch((const void*)(Af + g + 32), 0, 1);
      }
      *(u16x4*)&As[r][c4] = hv;
    }
    // stage W: 32x64, vec4 along n: 512 vecs -> 2 per thread
#pragma unroll
    for (int v = tid; v < 32 * 16; v += 256) {
      int r = v >> 4, c4 = (v & 15) * 4;
      size_t g = (size_t)(kk + r) * N + (n0 + c4);
      f32x4 f = *(const f32x4*)(Wt + g);
      if (kk + 32 < Dd) __builtin_prefetch((const void*)(Wt + g + 32 * (size_t)N), 0, 1);
      *(u16x4*)&Bs[r][c4] = f2bf4(f);
    }
    __syncthreads();
    Frag16 a0, a1, b0, b1;
    load_afrag(a0, &As[0][0], 36, wm * 32, lane);
    load_afrag(a1, &As[0][0], 36, wm * 32 + 16, lane);
    load_bfrag(b0, &Bs[0][0], 68, wn * 32, lane);
    load_bfrag(b1, &Bs[0][0], 68, wn * 32 + 16, lane);
    WMMA_BF16(acc00, a0, b0);
    WMMA_BF16(acc01, a0, b1);
    WMMA_BF16(acc10, a1, b0);
    WMMA_BF16(acc11, a1, b1);
    __syncthreads();
  }
  // C layout: lane&15 = N, VGPR j -> M = j + 8*(lane>>4)
  float* Cf = (float*)Cv;
  unsigned short* Cb = (unsigned short*)Cv;
#pragma unroll
  for (int mi = 0; mi < 2; ++mi) {
    v8f* arow0 = mi ? &acc10 : &acc00;
    v8f* arow1 = mi ? &acc11 : &acc01;
    int mb = m0 + wm * 32 + mi * 16 + ((lane >> 4) * 8);
#pragma unroll
    for (int j = 0; j < 8; ++j) {
      int gm = mb + j;
      int gn0 = n0 + wn * 32 + (lane & 15);
      int gn1 = gn0 + 16;
      float v0 = (*arow0)[j] + bias[gn0];
      float v1 = (*arow1)[j] + bias[gn1];
      if (OUT_BF16) {
        Cb[(size_t)gm * N + gn0] = f2bf(v0);
        Cb[(size_t)gm * N + gn1] = f2bf(v1);
      } else {
        Cf[(size_t)gm * N + gn0] = v0;
        Cf[(size_t)gm * N + gn1] = v1;
      }
    }
  }
}

// =====================================================================
// Kernel 2: scores E[b,h,q,k] = (q.k)/SCALE (+r), masked.
// block tile 128(Q) x 64(K); k-operand staged [64 kcols][32 dk], gathered transposed.
// grid: x = k-blocks(64), y = q-blocks(128) = 2, z = b*H + h
// =====================================================================
__global__ void __launch_bounds__(256)
gemm_scores_kernel(const unsigned short* __restrict__ Qp,
                   const unsigned short* __restrict__ Kp,
                   const int* __restrict__ mask, const float* __restrict__ r,
                   int add_r, float* __restrict__ E) {
  __shared__ __attribute__((aligned(16))) unsigned short Qs[128][36];
  __shared__ __attribute__((aligned(16))) unsigned short Ks[64][36];
  const int tid = threadIdx.x, lane = tid & 31, wave = tid >> 5;
  const int wm = wave & 3, wn = wave >> 2;
  const int bh = blockIdx.z, b = bh / H_SZ, h = bh % H_SZ;
  const int q0 = blockIdx.y * 128, k0 = blockIdx.x * 64;
  const size_t qbase = (size_t)b * Q_SZ * ADIM_SZ + (size_t)h * DK_SZ;
  const size_t kbase = (size_t)b * K_SZ * ADIM_SZ + (size_t)h * DK_SZ;

  v8f acc00 = {}; v8f acc01 = {}; v8f acc10 = {}; v8f acc11 = {};

  for (int dd = 0; dd < DK_SZ; dd += 32) {
    // Q tile: 128 x 32, guard-free (Q exact)
#pragma unroll
    for (int v = tid; v < 128 * 8; v += 256) {
      int rr = v >> 3, c4 = (v & 7) * 4;
      size_t g = qbase + (size_t)(q0 + rr) * ADIM_SZ + dd + c4;
      *(u16x4*)&Qs[rr][c4] = *(const u16x4*)(Qp + g);
      if (dd + 32 < DK_SZ) __builtin_prefetch((const void*)(Qp + g + 32), 0, 1);
    }
    // K tile: 64 rows (output cols, guarded branchlessly) x 32 dk
#pragma unroll
    for (int v = tid; v < 64 * 8; v += 256) {
      int rr = v >> 3, c4 = (v & 7) * 4;
      int gk = k0 + rr;
      int gks = (gk < K_SZ) ? gk : 0;
      u16x4 hv = *(const u16x4*)(Kp + kbase + (size_t)gks * ADIM_SZ + dd + c4);
      if (gk >= K_SZ) { u16x4 z = {0, 0, 0, 0}; hv = z; }
      *(u16x4*)&Ks[rr][c4] = hv;
    }
    __syncthreads();
    Frag16 a0, a1, b0, b1;
    load_afrag(a0, &Qs[0][0], 36, wm * 32, lane);
    load_afrag(a1, &Qs[0][0], 36, wm * 32 + 16, lane);
    load_bfrag_t(b0, &Ks[0][0], 36, wn * 32, lane);
    load_bfrag_t(b1, &Ks[0][0], 36, wn * 32 + 16, lane);
    WMMA_BF16(acc00, a0, b0);
    WMMA_BF16(acc01, a0, b1);
    WMMA_BF16(acc10, a1, b0);
    WMMA_BF16(acc11, a1, b1);
    __syncthreads();
  }
  float radd = add_r ? r[0] : 0.0f;
  size_t ebase = ((size_t)bh) * Q_SZ * K_SZ;
#pragma unroll
  for (int mi = 0; mi < 2; ++mi) {
    v8f* arow0 = mi ? &acc10 : &acc00;
    v8f* arow1 = mi ? &acc11 : &acc01;
    int qb = q0 + wm * 32 + mi * 16 + ((lane >> 4) * 8);
#pragma unroll
    for (int j = 0; j < 8; ++j) {
      int gq = qb + j;
      int gk0 = k0 + wn * 32 + (lane & 15);
      int gk1 = gk0 + 16;
      size_t mrow = ((size_t)b * Q_SZ + gq) * K_SZ;
      if (gk0 < K_SZ) {
        float v = (*arow0)[j] / SCALE_F + radd;
        if (mask[mrow + gk0] == 0) v = NEG_F;
        E[ebase + (size_t)gq * K_SZ + gk0] = v;
      }
      if (gk1 < K_SZ) {
        float v = (*arow1)[j] / SCALE_F + radd;
        if (mask[mrow + gk1] == 0) v = NEG_F;
        E[ebase + (size_t)gq * K_SZ + gk1] = v;
      }
    }
  }
}

// =====================================================================
// Kernel 3: p = sigmoid(e) in place; cp = exp(excl-cumsum(log clip(1-p,EPS,1)))
// =====================================================================
__global__ void __launch_bounds__(256)
sigmoid_cp_kernel(float* __restrict__ P, float* __restrict__ CP, int Kn) {
  __shared__ float s[256];
  const int tid = threadIdx.x;
  size_t base = (size_t)blockIdx.x * Kn;
  float carry = 0.0f;
  int nchunk = (Kn + 255) / 256;
  for (int c = 0; c < nchunk; ++c) {
    int k = c * 256 + tid;
    float l = 0.0f;
    if (k < Kn) {
      float e = P[base + k];
      float pv = 1.0f / (1.0f + __expf(-e));
      P[base + k] = pv;
      float om = fminf(fmaxf(1.0f - pv, EPS_F), 1.0f);
      l = __logf(om);
    }
    s[tid] = l;
    __syncthreads();
    for (int off = 1; off < 256; off <<= 1) {
      float t = 0.0f;
      if (tid >= off) t = s[tid - off];
      __syncthreads();
      s[tid] += t;
      __syncthreads();
    }
    float incl = s[tid];
    float tot = s[255];
    if (k < Kn) CP[base + k] = __expf(carry + (incl - l));
    __syncthreads();
    carry += tot;
  }
}

// =====================================================================
// Kernel 4: monotonic recurrence over q. One block per (b,h); aw in LDS;
// alpha written in-place over P.
// =====================================================================
__global__ void __launch_bounds__(256)
alpha_kernel(const float* __restrict__ aw_prev, const float* __restrict__ CP,
             float* __restrict__ PA, int Qn, int Kn) {
  __shared__ float aw[2048];
  __shared__ float s[256];
  const int tid = threadIdx.x;
  const int bh = blockIdx.x;
  for (int k = tid; k < Kn; k += 256) aw[k] = aw_prev[(size_t)bh * Kn + k];
  __syncthreads();
  int nchunk = (Kn + 255) / 256;
  for (int q = 0; q < Qn; ++q) {
    size_t rb = ((size_t)bh * Qn + q) * (size_t)Kn;
    float carry = 0.0f;
    for (int c = 0; c < nchunk; ++c) {
      int k = c * 256 + tid;
      float pk = 0.0f, cpk = 0.0f, y = 0.0f;
      if (k < Kn) {
        pk = PA[rb + k];
        cpk = CP[rb + k];
        float den = fminf(fmaxf(cpk, EPS_F), 1.0f);
        y = aw[k] / den;
      }
      s[tid] = y;
      __syncthreads();
      for (int off = 1; off < 256; off <<= 1) {
        float t = 0.0f;
        if (tid >= off) t = s[tid - off];
        __syncthreads();
        s[tid] += t;
        __syncthreads();
      }
      float incl = s[tid];
      float tot = s[255];
      float val = pk * cpk * (carry + incl);
      __syncthreads();
      if (k < Kn) { PA[rb + k] = val; aw[k] = val; }
      carry += tot;
    }
    __syncthreads();
  }
}

// =====================================================================
// Kernel 5: beta (window sums), one block per (b,h,q) row -> bf16
// =====================================================================
__global__ void __launch_bounds__(256)
beta_kernel(const float* __restrict__ E, const float* __restrict__ A,
            unsigned short* __restrict__ Bb, int Kn) {
  __shared__ float S[2048];
  __shared__ float G[2048];
  __shared__ float red[256];
  const int tid = threadIdx.x;
  size_t base = (size_t)blockIdx.x * Kn;
  float m = -3.4e38f;
  for (int k = tid; k < Kn; k += 256) m = fmaxf(m, E[base + k]);
  red[tid] = m;
  __syncthreads();
  for (int off = 128; off > 0; off >>= 1) {
    if (tid < off) red[tid] = fmaxf(red[tid], red[tid + off]);
    __syncthreads();
  }
  m = red[0];
  __syncthreads();
  for (int k = tid; k < Kn; k += 256)
    S[k] = fmaxf(__expf(E[base + k] - m), 1e-5f);
  __syncthreads();
  for (int k = tid; k < Kn; k += 256) {
    float d = 0.0f;
#pragma unroll
    for (int j = 0; j < WWIN; ++j) { int kk = k - j; if (kk >= 0) d += S[kk]; }
    G[k] = A[base + k] * SHARP_F / d;
  }
  __syncthreads();
  for (int k = tid; k < Kn; k += 256) {
    float g = 0.0f;
#pragma unroll
    for (int j = 0; j < WWIN; ++j) { int kk = k + j; if (kk < Kn) g += G[kk]; }
    Bb[base + k] = f2bf(S[k] * g);
  }
}

// =====================================================================
// Kernel 6: cv[b,q,h*dk+n] = sum_k beta[b,h,q,k] * v[b,k,h*dk+n]
// tile 128(Q) x 64(N); inner K=2000 zero-padded (K%4==0 -> vec4 all-in/out)
// grid: x = n-blocks(64)=4, y = q-blocks(128)=2, z = b*H+h
// =====================================================================
__global__ void __launch_bounds__(256)
gemm_cv_kernel(const unsigned short* __restrict__ Beta,
               const unsigned short* __restrict__ Vp,
               unsigned short* __restrict__ CV) {
  __shared__ __attribute__((aligned(16))) unsigned short As[128][36];
  __shared__ __attribute__((aligned(16))) unsigned short Bs[32][68];
  const int tid = threadIdx.x, lane = tid & 31, wave = tid >> 5;
  const int wm = wave & 3, wn = wave >> 2;
  const int bh = blockIdx.z, b = bh / H_SZ, h = bh % H_SZ;
  const int q0 = blockIdx.y * 128, n0 = blockIdx.x * 64;
  const size_t abase = (size_t)bh * Q_SZ * K_SZ;
  const size_t vbase = (size_t)b * K_SZ * ADIM_SZ + (size_t)h * DK_SZ;

  v8f acc00 = {}; v8f acc01 = {}; v8f acc10 = {}; v8f acc11 = {};
  const int ksteps = (K_SZ + 31) / 32;   // 63
  for (int t = 0; t < ksteps; ++t) {
    int kk = t * 32;
    // beta tile: 128 q-rows x 32 k (col-guarded, vec4 aligned since K%4==0)
#pragma unroll
    for (int v = tid; v < 128 * 8; v += 256) {
      int rr = v >> 3, c4 = (v & 7) * 4;
      int gk = kk + c4;
      int gks = (gk < K_SZ) ? gk : 0;
      u16x4 hv = *(const u16x4*)(Beta + abase + (size_t)(q0 + rr) * K_SZ + gks);
      if (gk >= K_SZ) { u16x4 z = {0, 0, 0, 0}; hv = z; }
      *(u16x4*)&As[rr][c4] = hv;
    }
    // v tile: 32 k-rows (guarded) x 64 n
#pragma unroll
    for (int v = tid; v < 32 * 16; v += 256) {
      int rr = v >> 4, c4 = (v & 15) * 4;
      int gk = kk + rr;
      int gks = (gk < K_SZ) ? gk : 0;
      u16x4 hv = *(const u16x4*)(Vp + vbase + (size_t)gks * ADIM_SZ + n0 + c4);
      if (gk >= K_SZ) { u16x4 z = {0, 0, 0, 0}; hv = z; }
      *(u16x4*)&Bs[rr][c4] = hv;
    }
    __syncthreads();
    Frag16 a0, a1, b0, b1;
    load_afrag(a0, &As[0][0], 36, wm * 32, lane);
    load_afrag(a1, &As[0][0], 36, wm * 32 + 16, lane);
    load_bfrag(b0, &Bs[0][0], 68, wn * 32, lane);
    load_bfrag(b1, &Bs[0][0], 68, wn * 32 + 16, lane);
    WMMA_BF16(acc00, a0, b0);
    WMMA_BF16(acc01, a0, b1);
    WMMA_BF16(acc10, a1, b0);
    WMMA_BF16(acc11, a1, b1);
    __syncthreads();
  }
#pragma unroll
  for (int mi = 0; mi < 2; ++mi) {
    v8f* arow0 = mi ? &acc10 : &acc00;
    v8f* arow1 = mi ? &acc11 : &acc01;
    int qb = q0 + wm * 32 + mi * 16 + ((lane >> 4) * 8);
#pragma unroll
    for (int j = 0; j < 8; ++j) {
      int gq = qb + j;
      int gn0 = n0 + wn * 32 + (lane & 15);
      size_t orow = ((size_t)b * Q_SZ + gq) * ADIM_SZ + (size_t)h * DK_SZ;
      CV[orow + gn0]      = f2bf((*arow0)[j]);
      CV[orow + gn0 + 16] = f2bf((*arow1)[j]);
    }
  }
}

// =====================================================================
// host-side orchestration
// =====================================================================
extern "C" void kernel_launch(void* const* d_in, const int* in_sizes, int n_in,
                              void* d_out, int out_size, void* d_ws, size_t ws_size,
                              hipStream_t stream) {
  (void)in_sizes; (void)n_in; (void)out_size; (void)ws_size;
  const float* key     = (const float*)d_in[0];
  const float* value   = (const float*)d_in[1];
  const float* query   = (const float*)d_in[2];
  const int*   mask    = (const int*)d_in[3];
  const float* aw_prev = (const float*)d_in[4];
  const float* Wk_ma   = (const float*)d_in[5];
  const float* bk_ma   = (const float*)d_in[6];
  const float* Wq_ma   = (const float*)d_in[7];
  const float* bq_ma   = (const float*)d_in[8];
  const float* rvec    = (const float*)d_in[9];
  const float* Wk_ca   = (const float*)d_in[10];
  const float* bk_ca   = (const float*)d_in[11];
  const float* Wq_ca   = (const float*)d_in[12];
  const float* bq_ca   = (const float*)d_in[13];
  const float* Wv      = (const float*)d_in[14];
  const float* bv      = (const float*)d_in[15];
  const float* Wo      = (const float*)d_in[16];
  const float* bo      = (const float*)d_in[17];
  float* out = (float*)d_out;

  const size_t MK = (size_t)B_SZ * K_SZ;          // 32000
  const size_t MQ = (size_t)B_SZ * Q_SZ;          // 4096
  const size_t NSCORE = (size_t)B_SZ * H_SZ * Q_SZ * K_SZ;

  char* ws = (char*)d_ws;
  size_t off = 0;
  auto carve = [&](size_t bytes) -> void* {
    void* p = ws + off;
    off += (bytes + 255) & ~(size_t)255;
    return p;
  };
  unsigned short* kma  = (unsigned short*)carve(MK * ADIM_SZ * 2);
  unsigned short* kca  = (unsigned short*)carve(MK * ADIM_SZ * 2);
  unsigned short* vbf  = (unsigned short*)carve(MK * ADIM_SZ * 2);
  unsigned short* qma  = (unsigned short*)carve(MQ * ADIM_SZ * 2);
  unsigned short* qca  = (unsigned short*)carve(MQ * ADIM_SZ * 2);
  float* pbuf          = (float*)carve(NSCORE * 4);           // e_ma -> p -> alpha
  float* cpbuf         = (float*)carve(NSCORE * 4);
  float* eca           = (float*)carve(NSCORE * 4);
  unsigned short* bbf  = (unsigned short*)carve(NSCORE * 2);
  unsigned short* cvbf = (unsigned short*)carve(MQ * ADIM_SZ * 2);

  dim3 blk(256, 1, 1);
  dim3 gK((unsigned)(MK / 128), ADIM_SZ / 64, 1);   // (250,16)
  dim3 gQ((unsigned)(MQ / 128), ADIM_SZ / 64, 1);   // (32,16)

  // projections (f32 in -> bf16 out)
  gemm_proj_kernel<false, true><<<gK, blk, 0, stream>>>(key,   Wk_ma, bk_ma, kma, (int)MK, D_SZ, ADIM_SZ);
  gemm_proj_kernel<false, true><<<gQ, blk, 0, stream>>>(query, Wq_ma, bq_ma, qma, (int)MQ, D_SZ, ADIM_SZ);
  gemm_proj_kernel<false, true><<<gK, blk, 0, stream>>>(key,   Wk_ca, bk_ca, kca, (int)MK, D_SZ, ADIM_SZ);
  gemm_proj_kernel<false, true><<<gQ, blk, 0, stream>>>(query, Wq_ca, bq_ca, qca, (int)MQ, D_SZ, ADIM_SZ);
  gemm_proj_kernel<false, true><<<gK, blk, 0, stream>>>(value, Wv,    bv,    vbf, (int)MK, D_SZ, ADIM_SZ);

  // attention scores
  dim3 gS((K_SZ + 63) / 64, Q_SZ / 128, B_SZ * H_SZ);   // (32,2,64)
  gemm_scores_kernel<<<gS, blk, 0, stream>>>(qma, kma, mask, rvec, 1, pbuf);
  gemm_scores_kernel<<<gS, blk, 0, stream>>>(qca, kca, mask, rvec, 0, eca);

  // p / cp
  sigmoid_cp_kernel<<<(unsigned)(B_SZ * H_SZ * Q_SZ), blk, 0, stream>>>(pbuf, cpbuf, K_SZ);

  // monotonic alpha recurrence (alpha overwrites pbuf)
  alpha_kernel<<<B_SZ * H_SZ, blk, 0, stream>>>(aw_prev, cpbuf, pbuf, Q_SZ, K_SZ);

  // chunkwise beta
  beta_kernel<<<(unsigned)(B_SZ * H_SZ * Q_SZ), blk, 0, stream>>>(eca, pbuf, bbf, K_SZ);

  // context vectors
  dim3 gC(DK_SZ / 64, Q_SZ / 128, B_SZ * H_SZ);         // (4,2,64)
  gemm_cv_kernel<<<gC, blk, 0, stream>>>(bbf, vbf, cvbf);

  // output projection (bf16 A -> f32 out)
  gemm_proj_kernel<true, false><<<gQ, blk, 0, stream>>>(cvbf, Wo, bo, out, (int)MQ, ADIM_SZ, D_SZ);
}
